// CausalAttention_41154376630798
// MI455X (gfx1250) — matrix operand order repaired and checked by reading, not compiled, not run on hardware
//
#include <hip/hip_runtime.h>
#include <hip/hip_bf16.h>

typedef __bf16 bf16;
typedef __attribute__((ext_vector_type(16))) __bf16 v16bf;
typedef __attribute__((ext_vector_type(8)))  __bf16 v8bf;
typedef __attribute__((ext_vector_type(4)))  __bf16 v4bf;
typedef __attribute__((ext_vector_type(8)))  float  v8f;

#define T_SEQ   4096
#define DMODEL  1024
#define DK      256
#define NB      4

static __device__ inline v16bf cat8(v8bf lo, v8bf hi) {
  return __builtin_shufflevector(lo, hi, 0,1,2,3,4,5,6,7,8,9,10,11,12,13,14,15);
}

static __device__ inline v8f wmma_bf16(v16bf a, v16bf b, v8f c) {
  return __builtin_amdgcn_wmma_f32_16x16x32_bf16(false, a, false, b, (short)0, c, false, false);
}

// ---- CDNA5 async Global->LDS copy (ASYNCcnt), guarded fallback ----
#if __has_builtin(__builtin_amdgcn_global_load_async_to_lds_b128)
#define HAS_ASYNC_LDS 1
typedef int v4i __attribute__((vector_size(16)));
typedef __attribute__((address_space(1))) v4i as1_v4i;
typedef __attribute__((address_space(3))) v4i as3_v4i;
static __device__ inline void async_b128(const void* g, void* l) {
  __builtin_amdgcn_global_load_async_to_lds_b128(
      (as1_v4i*)g, (as3_v4i*)l, 0, 0);
}
#else
#define HAS_ASYNC_LDS 0
#endif

static __device__ inline void wait_async0() {
#if __has_builtin(__builtin_amdgcn_s_wait_asynccnt)
  __builtin_amdgcn_s_wait_asynccnt(0);
#else
  asm volatile("s_wait_asynccnt 0" ::: "memory");
#endif
}

// ---------------------------------------------------------------------------
// Kernel 0a: W [1024,256] f32 -> Wt [proj][256][1024] bf16 (col-major panels)
// ---------------------------------------------------------------------------
__global__ __launch_bounds__(256) void wt_convert(
    const float* __restrict__ Wq, const float* __restrict__ Wk,
    const float* __restrict__ Wv, bf16* __restrict__ Wt)
{
  int idx = blockIdx.x * 256 + threadIdx.x;   // 0 .. 3*256*1024-1
  int p   = idx >> 18;
  int rem = idx & 262143;
  int n   = rem >> 10;
  int k   = rem & 1023;
  const float* W = (p == 0) ? Wq : ((p == 1) ? Wk : Wv);
  Wt[idx] = (bf16)W[k * DK + n];
}

// ---------------------------------------------------------------------------
// Kernel 0b: X f32 [16384,1024] -> Xb bf16 (read X from HBM exactly once)
// ---------------------------------------------------------------------------
__global__ __launch_bounds__(256) void x_convert(
    const float* __restrict__ X, bf16* __restrict__ Xb)
{
  int i = (blockIdx.x * 256 + threadIdx.x) * 4;
  float4 f = *(const float4*)(X + i);
  v4bf o;
  o[0] = (bf16)f.x; o[1] = (bf16)f.y; o[2] = (bf16)f.z; o[3] = (bf16)f.w;
  *(v4bf*)(Xb + i) = o;
}

// ---------------------------------------------------------------------------
// Kernel 1: fused QKV projection. One block (8 waves) per 16-row m-tile covers
// all 768 output columns (48 n-tiles, 6 per wave) -> Xb read once from HBM.
// Q,K stored bf16 [B*T][256]; V stored transposed bf16 [B][256][T].
// ---------------------------------------------------------------------------
__global__ __launch_bounds__(256) void qkv_proj(
    const bf16* __restrict__ Xb, const bf16* __restrict__ Wt,
    const float* __restrict__ bq, const float* __restrict__ bk,
    const float* __restrict__ bv,
    bf16* __restrict__ Qg, bf16* __restrict__ Kg, bf16* __restrict__ Vt)
{
  const int lane = threadIdx.x & 31;
  const int wave = threadIdx.x >> 5;
  const int ln   = lane & 15;
  const int h    = lane >> 4;
  const int mtile = blockIdx.x;   // 0..1023 (16 rows each)

  const v8bf* xrow = (const v8bf*)(Xb + (size_t)(mtile * 16 + ln) * DMODEL);

  // wave handles flattened n-tiles nt = wave*6 + i  (768 cols = 48 tiles)
  const v8bf* wrow[6];
  #pragma unroll
  for (int i = 0; i < 6; ++i) {
    int nt = wave * 6 + i;
    wrow[i] = (const v8bf*)(Wt + ((size_t)(nt * 16 + ln)) * DMODEL);
  }

  v8f acc[6];
  #pragma unroll
  for (int i = 0; i < 6; ++i) acc[i] = {};

  #pragma unroll 2
  for (int kc = 0; kc < 32; ++kc) {
    v16bf a = cat8(xrow[kc * 4 + 2 * h], xrow[kc * 4 + 2 * h + 1]);   // A: 16x32 tile
    #pragma unroll
    for (int i = 0; i < 6; ++i) {
      v16bf bm = cat8(wrow[i][kc * 4 + 2 * h], wrow[i][kc * 4 + 2 * h + 1]);
      acc[i] = wmma_bf16(a, bm, acc[i]);
    }
  }

  #pragma unroll
  for (int i = 0; i < 6; ++i) {
    const int nt    = wave * 6 + i;
    const int proj  = nt >> 4;          // 0:Q 1:K 2:V
    const int nbase = (nt & 15) * 16;
    const float* bias = (proj == 0) ? bq : ((proj == 1) ? bk : bv);
    const float bn = bias[nbase + ln];
    if (proj == 2) {
      #pragma unroll
      for (int r = 0; r < 8; ++r) {
        int m = mtile * 16 + r + 8 * h;           // global row (b*T + t)
        int bb = m >> 12, t = m & (T_SEQ - 1);
        Vt[((size_t)(bb * DK + nbase + ln)) * T_SEQ + t] = (bf16)(acc[i][r] + bn);
      }
    } else {
      bf16* outp = (proj == 0) ? Qg : Kg;
      #pragma unroll
      for (int r = 0; r < 8; ++r) {
        int m = mtile * 16 + r + 8 * h;
        outp[(size_t)m * DK + nbase + ln] = (bf16)(acc[i][r] + bn);
      }
    }
  }
}

// ---------------------------------------------------------------------------
// Kernel 2: fused causal flash attention, software-pipelined async DMA.
// Block = 128 threads = 4 waves; each wave owns 16 query rows.
// K_s is prefetched during step s-1's P*V; V_s is fetched during step s's
// score WMMAs. Single-buffered LDS, two barriers per step.
// ---------------------------------------------------------------------------
__global__ __launch_bounds__(128) void flash_attn(
    const bf16* __restrict__ Qg, const bf16* __restrict__ Kg,
    const bf16* __restrict__ Vtg, float* __restrict__ out)
{
  __shared__ __align__(16) bf16 K_lds[32 * DK];     // [s][d] 16KB
  __shared__ __align__(16) bf16 V_lds[DK * 32];     // [d][s] 16KB
  __shared__ __align__(16) bf16 P_lds[4 * 16 * 32]; // per-wave P staging 4KB

  const int tid  = threadIdx.x;
  const int lane = tid & 31;
  const int wave = tid >> 5;
  const int ln   = lane & 15;
  const int h    = lane >> 4;

  const int bidx  = blockIdx.x >> 6;   // batch
  const int qbase = (blockIdx.x & 63) * 64;
  const int tbase = qbase + wave * 16; // this wave's 16 query rows

  // tile loaders (all 128 threads participate)
  auto issue_K = [&](int sb) {
#if HAS_ASYNC_LDS
    const char* kg = (const char*)(Kg + ((size_t)(bidx * T_SEQ + sb)) * DK);
    char* kl = (char*)K_lds;
    #pragma unroll
    for (int i = 0; i < 8; ++i) {                    // 1024 x 16B
      int off = (tid + 128 * i) * 16;
      async_b128(kg + off, kl + off);
    }
#else
    const uint4* src = (const uint4*)(Kg + ((size_t)(bidx * T_SEQ + sb)) * DK);
    uint4* dst = (uint4*)K_lds;
    #pragma unroll
    for (int i = 0; i < 8; ++i) dst[tid + 128 * i] = src[tid + 128 * i];
#endif
  };
  auto issue_V = [&](int sb) {
#if HAS_ASYNC_LDS
    #pragma unroll
    for (int i = 0; i < 2; ++i) {                    // 256 rows x 64B
      int d = tid + 128 * i;
      const char* vg = (const char*)(Vtg + ((size_t)(bidx * DK + d)) * T_SEQ + sb);
      char* vl = (char*)V_lds + d * 64;
      #pragma unroll
      for (int j = 0; j < 4; ++j) async_b128(vg + j * 16, vl + j * 16);
    }
#else
    #pragma unroll
    for (int i = 0; i < 2; ++i) {
      int d = tid + 128 * i;
      const uint4* vs = (const uint4*)(Vtg + ((size_t)(bidx * DK + d)) * T_SEQ + sb);
      uint4* vd = (uint4*)(V_lds + d * 32);
      vd[0] = vs[0]; vd[1] = vs[1]; vd[2] = vs[2]; vd[3] = vs[3];
    }
#endif
  };

  // Preload Q A-operands: 8 chunks of 16x32, kept in 64 VGPRs.
  const v8bf* qrow = (const v8bf*)(Qg + ((size_t)(bidx * T_SEQ + tbase + ln)) * DK);
  v16bf qreg[8];
  #pragma unroll
  for (int kc = 0; kc < 8; ++kc)
    qreg[kc] = cat8(qrow[kc * 4 + 2 * h], qrow[kc * 4 + 2 * h + 1]);

  v8f acc[16];
  #pragma unroll
  for (int n = 0; n < 16; ++n) acc[n] = {};
  float mrow[8], lrow[8];
  #pragma unroll
  for (int r = 0; r < 8; ++r) { mrow[r] = -3.0e38f; lrow[r] = 0.0f; }

  bf16* Pw = P_lds + wave * 16 * 32;
  const int nsteps = (qbase + 64) >> 5;   // keys [0, qbase+64) step 32

  issue_K(0);                             // pipeline prologue

  for (int step = 0; step < nsteps; ++step) {
    const int sbase = step * 32;
    const bool live = (sbase <= tbase + 15);   // wave-uniform causal skip

    wait_async0();        // K_s resident (this thread's chunks)
    __syncthreads();      // ... and everyone's; V_{s-1} readers done
    issue_V(sbase);       // V_s DMA overlaps score WMMAs below

    v8f s0 = {}, s1 = {};
    float p0[8], p1[8], corr[8];
    if (live) {
      // ---- scores: S = Q K^T, two 16x16 subtiles over k=256 ----
      const v8bf* k0 = (const v8bf*)(K_lds + (size_t)ln * DK);
      const v8bf* k1 = (const v8bf*)(K_lds + (size_t)(16 + ln) * DK);
      #pragma unroll
      for (int kc = 0; kc < 8; ++kc) {
        v16bf b0 = cat8(k0[kc * 4 + 2 * h], k0[kc * 4 + 2 * h + 1]);
        v16bf b1 = cat8(k1[kc * 4 + 2 * h], k1[kc * 4 + 2 * h + 1]);
        s0 = wmma_bf16(qreg[kc], b0, s0);
        s1 = wmma_bf16(qreg[kc], b1, s1);
      }

      // ---- mask (before scale, as reference) + online softmax ----
      const float scale = 0.0625f;           // 1/sqrt(256)
      const int c0 = sbase + ln, c1 = sbase + 16 + ln;
      #pragma unroll
      for (int r = 0; r < 8; ++r) {
        const int t = tbase + r + 8 * h;
        float a0 = (c0 <= t) ? s0[r] * scale : -6.25e7f;   // -1e9/16
        float a1 = (c1 <= t) ? s1[r] * scale : -6.25e7f;
        float mx = fmaxf(a0, a1);
        #pragma unroll
        for (int msk = 1; msk < 16; msk <<= 1)
          mx = fmaxf(mx, __shfl_xor(mx, msk, 32));
        float mnew = fmaxf(mrow[r], mx);
        corr[r] = __expf(mrow[r] - mnew);
        mrow[r] = mnew;
        p0[r] = __expf(a0 - mnew);
        p1[r] = __expf(a1 - mnew);
        float sum = p0[r] + p1[r];
        #pragma unroll
        for (int msk = 1; msk < 16; msk <<= 1)
          sum += __shfl_xor(sum, msk, 32);
        lrow[r] = lrow[r] * corr[r] + sum;
      }
      #pragma unroll
      for (int n = 0; n < 16; ++n)
        #pragma unroll
        for (int r = 0; r < 8; ++r)
          acc[n][r] *= corr[r];
    }

    wait_async0();        // V_s resident (this thread's chunks)
    __syncthreads();      // ... and everyone's; K_s readers (scores) done
    if (step + 1 < nsteps)
      issue_K(sbase + 32);   // K_{s+1} DMA overlaps P-staging + P*V below

    if (live) {
      // ---- C-layout P -> A-layout via per-wave LDS staging ----
      #pragma unroll
      for (int r = 0; r < 8; ++r) {
        Pw[(r + 8 * h) * 32 + ln]      = (bf16)p0[r];
        Pw[(r + 8 * h) * 32 + 16 + ln] = (bf16)p1[r];
      }
      asm volatile("s_wait_dscnt 0" ::: "memory");
      const v8bf* prow = (const v8bf*)(Pw + ln * 32);
      v16bf pa = cat8(prow[2 * h], prow[2 * h + 1]);

      // ---- O += P V  (16 N-tiles of 16 columns) ----
      #pragma unroll
      for (int n = 0; n < 16; ++n) {
        const v8bf* vrow = (const v8bf*)(V_lds + (size_t)(n * 16 + ln) * 32);
        v16bf bm = cat8(vrow[2 * h], vrow[2 * h + 1]);
        acc[n] = wmma_bf16(pa, bm, acc[n]);
      }
    }
  }

  // ---- epilogue: O / l, fp32 out [B][T][256] ----
  float rl[8];
  #pragma unroll
  for (int r = 0; r < 8; ++r) rl[r] = 1.0f / lrow[r];
  #pragma unroll
  for (int n = 0; n < 16; ++n) {
    #pragma unroll
    for (int r = 0; r < 8; ++r) {
      const int t = tbase + r + 8 * h;
      out[((size_t)(bidx * T_SEQ + t)) * DK + n * 16 + ln] = acc[n][r] * rl[r];
    }
  }
}

// ---------------------------------------------------------------------------
extern "C" void kernel_launch(void* const* d_in, const int* in_sizes, int n_in,
                              void* d_out, int out_size, void* d_ws, size_t ws_size,
                              hipStream_t stream)
{
  const float* X  = (const float*)d_in[0];
  const float* Wq = (const float*)d_in[1];
  const float* bq = (const float*)d_in[2];
  const float* Wk = (const float*)d_in[3];
  const float* bk = (const float*)d_in[4];
  const float* Wv = (const float*)d_in[5];
  const float* bv = (const float*)d_in[6];
  float* out = (float*)d_out;

  // workspace layout (bytes): Wt 1.5MB | Xb 32MB | Q 8MB | K 8MB | Vt 8MB
  char* ws = (char*)d_ws;
  bf16* Wt = (bf16*)(ws);
  bf16* Xb = (bf16*)(ws + 1572864);
  bf16* Qg = (bf16*)(ws + 1572864 + 33554432);
  bf16* Kg = (bf16*)(ws + 1572864 + 33554432 + 8388608);
  bf16* Vt = (bf16*)(ws + 1572864 + 33554432 + 2 * 8388608);

  wt_convert<<<dim3(3072), dim3(256), 0, stream>>>(Wq, Wk, Wv, Wt);
  x_convert<<<dim3(16384), dim3(256), 0, stream>>>(X, Xb);
  qkv_proj<<<dim3(1024), dim3(256), 0, stream>>>(Xb, Wt, bq, bk, bv, Qg, Kg, Vt);
  flash_attn<<<dim3(NB * (T_SEQ / 64)), dim3(128), 0, stream>>>(Qg, Kg, Vt, out);
}
